// CRF_88562225644114
// MI455X (gfx1250) — matrix operand order repaired and checked by reading, not compiled, not run on hardware
//
#include <hip/hip_runtime.h>
#include <hip/hip_bf16.h>

typedef __attribute__((ext_vector_type(16))) __bf16 v16bf;
typedef __attribute__((ext_vector_type(8)))  float  v8f;

#define BATCH 1024
#define SEQ   512
#define DIM   128
#define LAB   26
#define LPAD  32

__device__ __forceinline__ __bf16 f2bf(float f) { return (__bf16)f; }

// ---------------------------------------------------------------------------
// Kernel 1: emissions = X @ W via bf16 WMMA. Each wave computes a 16(rows of
// B*S) x 32(labels, padded) tile. K=128 -> 4 chained wmma per 16x16 N-tile.
// ---------------------------------------------------------------------------
__global__ __launch_bounds__(256) void emis_wmma_kernel(
    const float* __restrict__ X, const float* __restrict__ W,
    float* __restrict__ E) {
  const int lane  = threadIdx.x & 31;
  const int wave  = threadIdx.x >> 5;
  const int mtile = blockIdx.x * 8 + wave;          // 32768 tiles of 16 rows
  const long rowbase = (long)mtile * 16;

  const int lhalf = lane >> 4;                      // 0 or 1
  const int lmod  = lane & 15;

  // B-matrix (W, 128x32 padded) in registers: [ntile][kchunk]
  // Layout: N = lane%16; halves h -> K = kc*32 + (lane/16)*8 + h (+8 for h >= 8)
  v16bf Bm[2][4];
#pragma unroll
  for (int nt = 0; nt < 2; ++nt) {
    const int n = nt * 16 + lmod;
#pragma unroll
    for (int kc = 0; kc < 4; ++kc) {
      const int kbase = kc * 32 + lhalf * 8;
#pragma unroll
      for (int h = 0; h < 16; ++h) {
        float w = (n < LAB) ? W[(kbase + h + (h & 8)) * LAB + n] : 0.0f;
        Bm[nt][kc][h] = f2bf(w);
      }
    }
  }

  // A-matrix rows: lane (and lane+16) share row m = lane%16, disjoint K runs.
  const float* xrow = X + (rowbase + lmod) * (long)DIM;
  const int lo = lhalf * 8;                         // K run start within chunk

  v8f c0 = {}; v8f c1 = {};
#pragma unroll
  for (int kc = 0; kc < 4; ++kc) {
    const int k0 = kc * 32;
    const float4* p = reinterpret_cast<const float4*>(xrow + k0 + lo);
    const float4* q = reinterpret_cast<const float4*>(xrow + k0 + 16 + lo);
    float4 a0 = p[0], a1 = p[1];                    // K = k0+lo .. +7
    float4 b0 = q[0], b1 = q[1];                    // K = k0+16+lo .. +7
    v16bf A;
    A[0]  = f2bf(a0.x); A[1]  = f2bf(a0.y); A[2]  = f2bf(a0.z); A[3]  = f2bf(a0.w);
    A[4]  = f2bf(a1.x); A[5]  = f2bf(a1.y); A[6]  = f2bf(a1.z); A[7]  = f2bf(a1.w);
    A[8]  = f2bf(b0.x); A[9]  = f2bf(b0.y); A[10] = f2bf(b0.z); A[11] = f2bf(b0.w);
    A[12] = f2bf(b1.x); A[13] = f2bf(b1.y); A[14] = f2bf(b1.z); A[15] = f2bf(b1.w);
    c0 = __builtin_amdgcn_wmma_f32_16x16x32_bf16(false, A, false, Bm[0][kc],
                                                 (short)0, c0, false, false);
    c1 = __builtin_amdgcn_wmma_f32_16x16x32_bf16(false, A, false, Bm[1][kc],
                                                 (short)0, c1, false, false);
  }

  // C/D layout: VGPR j -> M = j + 8*(lane/16), N = lane%16 (+16 for c1)
  float* erow = E + (rowbase + 8 * lhalf) * (long)LPAD;
#pragma unroll
  for (int j = 0; j < 8; ++j) {
    erow[(long)j * LPAD + lmod]       = c0[j];
    erow[(long)j * LPAD + 16 + lmod]  = c1[j];
  }
}

// ---------------------------------------------------------------------------
// Kernel 2: Viterbi forward. One wave per batch; lane y holds delta[y].
// T columns in registers; broadcast delta[y'] via uniform-lane shfl
// (lowers to v_readlane -> scalar-operand VALU). Backpointers as bytes.
// ---------------------------------------------------------------------------
__global__ __launch_bounds__(256) void viterbi_fwd_kernel(
    const float* __restrict__ E, const float* __restrict__ T,
    unsigned char* __restrict__ BP, int* __restrict__ LAST) {
  const int lane = threadIdx.x & 31;
  const int wave = threadIdx.x >> 5;
  const int b    = blockIdx.x * 8 + wave;

  float Tc[LAB];
#pragma unroll
  for (int yp = 0; yp < LAB; ++yp)
    Tc[yp] = (lane < LAB) ? T[yp * LAB + lane] : -1e30f;

  const float* eb      = E  + (long)b * SEQ * LPAD;
  unsigned char* bpb   = BP + (long)b * SEQ * LPAD;

  float delta = (lane < LAB) ? eb[lane] : -1e30f;

  for (int s = 1; s < SEQ; ++s) {
    float e = eb[s * LPAD + lane];
    float best = -3.4e38f;
    int bidx = 0;
#pragma unroll
    for (int yp = 0; yp < LAB; ++yp) {
      float v = __shfl(delta, yp, 32) + Tc[yp];
      if (v > best) { best = v; bidx = yp; }       // first-max on ties (argmax)
    }
    delta = best + e;
    bpb[s * LPAD + lane] = (unsigned char)bidx;
  }

  // argmax over lanes (prefer lower index on ties)
  float bv = (lane < LAB) ? delta : -3.4e38f;
  int   bi = lane;
#pragma unroll
  for (int off = 16; off >= 1; off >>= 1) {
    float ov = __shfl_xor(bv, off, 32);
    int   oi = __shfl_xor(bi, off, 32);
    if (ov > bv || (ov == bv && oi < bi)) { bv = ov; bi = oi; }
  }
  if (lane == 0) LAST[b] = bi;
}

// ---------------------------------------------------------------------------
// Kernel 3: zero the one-hot output (float4 grid-stride stores).
// ---------------------------------------------------------------------------
__global__ void zero_out_kernel(float4* __restrict__ out, long n4) {
  long i      = (long)blockIdx.x * blockDim.x + threadIdx.x;
  long stride = (long)gridDim.x * blockDim.x;
  float4 z = {0.f, 0.f, 0.f, 0.f};
  for (; i < n4; i += stride) out[i] = z;
}

// ---------------------------------------------------------------------------
// Kernel 4: backtrace + scatter 1.0f. One thread per batch.
// ---------------------------------------------------------------------------
__global__ void backtrace_kernel(const unsigned char* __restrict__ BP,
                                 const int* __restrict__ LAST,
                                 float* __restrict__ out) {
  int b = blockIdx.x * blockDim.x + threadIdx.x;
  if (b >= BATCH) return;
  const unsigned char* bpb = BP + (long)b * SEQ * LPAD;
  float* ob = out + (long)b * SEQ * LAB;
  int y = LAST[b];
  ob[(SEQ - 1) * LAB + y] = 1.0f;
  for (int s = SEQ - 1; s >= 1; --s) {
    y = bpb[s * LPAD + y];
    ob[(s - 1) * LAB + y] = 1.0f;
  }
}

// ---------------------------------------------------------------------------
extern "C" void kernel_launch(void* const* d_in, const int* in_sizes, int n_in,
                              void* d_out, int out_size, void* d_ws,
                              size_t ws_size, hipStream_t stream) {
  const float* X = (const float*)d_in[0];   // [B,S,D] f32
  const float* W = (const float*)d_in[1];   // [D,L]   f32
  const float* T = (const float*)d_in[2];   // [L,L]   f32
  float* out = (float*)d_out;               // [B,S,L] f32 one-hot

  char* ws = (char*)d_ws;
  const size_t e_bytes  = (size_t)BATCH * SEQ * LPAD * sizeof(float); // 64 MiB
  const size_t bp_bytes = (size_t)BATCH * SEQ * LPAD;                 // 16 MiB
  float*         E    = (float*)ws;
  unsigned char* BP   = (unsigned char*)(ws + e_bytes);
  int*           LAST = (int*)(ws + e_bytes + bp_bytes);

  // 1) emissions via WMMA: 32768 M-tiles, 8 waves/block
  emis_wmma_kernel<<<(BATCH * SEQ / 16) / 8, 256, 0, stream>>>(X, W, E);
  // 2) Viterbi forward: 1024 waves, 8 waves/block
  viterbi_fwd_kernel<<<BATCH / 8, 256, 0, stream>>>(E, T, BP, LAST);
  // 3) zero output
  zero_out_kernel<<<2048, 256, 0, stream>>>((float4*)out, (long)out_size / 4);
  // 4) backtrace + scatter
  backtrace_kernel<<<(BATCH + 255) / 256, 256, 0, stream>>>(BP, LAST, out);
}
